// MLPAttention_6957847019898
// MI455X (gfx1250) — matrix operand order repaired and checked by reading, not compiled, not run on hardware
//
#include <hip/hip_runtime.h>

typedef unsigned short u16;
typedef unsigned int   u32;
typedef __bf16 bf16;
typedef bf16  v16bf __attribute__((ext_vector_type(16)));
typedef float v8f   __attribute__((ext_vector_type(8)));
typedef int   i32x4 __attribute__((ext_vector_type(4)));

constexpr int B_ = 2, L_ = 2048, DIM_ = 1024, H_ = 16, DH_ = 64, HID_ = 128;

#define WMMA_BF16(a, b, c) \
  __builtin_amdgcn_wmma_f32_16x16x32_bf16(false, (a), false, (b), (short)0, (c), false, false)

#if __has_builtin(__builtin_amdgcn_global_load_async_to_lds_b128)
#define HAVE_ASYNC_LDS 1
#else
#define HAVE_ASYNC_LDS 0
#endif

// ---------------- helpers ----------------

__device__ __forceinline__ u16 f2bf(float f) {
  u32 x = __float_as_uint(f);
  if ((x & 0x7fffffffu) > 0x7f800000u) return (u16)((x >> 16) | 0x40u);  // quiet NaN
  u32 r = x + 0x7fffu + ((x >> 16) & 1u);                                // RNE
  return (u16)(r >> 16);
}

union Frag16 { u32 u[8]; v16bf v; };

// A-fragment: 16x32 bf16 tile, row-major, base at (row0,k0), ld in elements (even).
// lane&15 = M; lanes<16 hold K 0..15, lanes>=16 hold K 16..31.
__device__ __forceinline__ v16bf load_a_frag(const u16* base, int ld, int lane) {
  Frag16 f;
  const u32* p = (const u32*)base;
  int m = lane & 15;
  int koff = (lane >> 4) << 4;  // 0 or 16
  int ro = m * ld;
#pragma unroll
  for (int i = 0; i < 8; ++i) {
    f.u[i]     = p[(ro + koff + 2 * i) >> 1];
  }
  return f.v;
}

// B-fragment from TRANSPOSED operand Bt[n][k] (row n = output column, K contiguous).
// lanes 0-15 hold K=0..15, lanes 16-31 hold K=16..31, 2 bf16/VGPR.
__device__ __forceinline__ v16bf load_b_frag(const u16* baseT, int ldk, int lane) {
  Frag16 f;
  const u32* p = (const u32*)baseT;
  int n = lane & 15;
  int kb = (lane >> 4) << 4;  // 0 or 16
  int ro = n * ldk;
#pragma unroll
  for (int i = 0; i < 8; ++i) f.u[i] = p[(ro + kb + 2 * i) >> 1];
  return f.v;
}

#if HAVE_ASYNC_LDS
__device__ __forceinline__ void async_cp16(const u16* g, u16* l) {
  __builtin_amdgcn_global_load_async_to_lds_b128(
      (__attribute__((address_space(1))) i32x4*)(unsigned long long)g,
      (__attribute__((address_space(3))) i32x4*)(unsigned int)(unsigned long long)l,
      0, 0);
}
__device__ __forceinline__ void wait_async0() {
#if __has_builtin(__builtin_amdgcn_s_wait_asynccnt)
  __builtin_amdgcn_s_wait_asynccnt(0);
#else
  asm volatile("s_wait_asynccnt 0x0" ::: "memory");
#endif
}
// stage V^T block: 64 rows x 32 cols bf16 (row stride L_) -> LDS tile ld=32
__device__ __forceinline__ void stage_v(const u16* Vb_s0, u16* dst, int lane) {
#pragma unroll
  for (int t = 0; t < 8; ++t) {
    int c = lane + 32 * t;      // 256 chunks of 16B
    int row = c >> 2, cc = c & 3;
    async_cp16(Vb_s0 + (size_t)row * L_ + cc * 8, dst + row * 32 + cc * 8);
  }
}
#endif

// ---------------- conversion kernels ----------------

__global__ void k_f32_to_bf16(const float* __restrict__ in, u16* __restrict__ out, int n) {
  int i = blockIdx.x * blockDim.x + threadIdx.x;
  if (i < n) out[i] = f2bf(in[i]);
}

// in: [K,N] f32 row-major  ->  out: [N,K] bf16 (transposed, K contiguous)
__global__ void k_transpose_bf16(const float* __restrict__ in, u16* __restrict__ out,
                                 int K, int N) {
  int i = blockIdx.x * blockDim.x + threadIdx.x;
  if (i < K * N) {
    int k = i / N, n = i % N;
    out[(size_t)n * K + k] = f2bf(in[i]);
  }
}

// ---------------- WMMA GEMM: one wave computes a 64x64 tile ----------------
// MODE 0: C = A@B+bias, store bf16 head-split  [B,H,L,DH]   (Xh)
// MODE 1: C = A@B+bias, store bf16 transposed  [B,H,DH,L]   (V^T)
// MODE 2: C = relu(A@B+bias), batched over z, store bf16 [z,M,N]  (h)
template <int MODE>
__global__ void __launch_bounds__(32)
gemm_wmma_k(const u16* __restrict__ A, const u16* __restrict__ Bt,
            const float* __restrict__ bias, u16* __restrict__ outp,
            int M, int N, int K, size_t strideA, size_t strideO) {
  int lane = threadIdx.x;
  int nt = blockIdx.x, mb = blockIdx.y, z = blockIdx.z;
  const u16* Ab = A + (size_t)z * strideA + (size_t)mb * 64 * K;
  const u16* Bb = Bt + (size_t)nt * 64 * K;

  v8f acc[4][4] = {};
  for (int k0 = 0; k0 < K; k0 += 32) {
    if (k0 + 32 < K) {  // gfx1250 global_prefetch_b8 for next K-slab
      __builtin_prefetch(Ab + (size_t)(lane & 15) * K + k0 + 32);
      __builtin_prefetch(Bb + (size_t)(lane & 15) * K + k0 + 32);
    }
    v16bf bfr[4], afr[4];
#pragma unroll
    for (int j = 0; j < 4; ++j)
      bfr[j] = load_b_frag(Bb + (size_t)(j * 16) * K + k0, K, lane);
#pragma unroll
    for (int i = 0; i < 4; ++i)
      afr[i] = load_a_frag(Ab + (size_t)(i * 16) * K + k0, K, lane);
#pragma unroll
    for (int j = 0; j < 4; ++j)
#pragma unroll
      for (int i = 0; i < 4; ++i)
        acc[j][i] = WMMA_BF16(afr[i], bfr[j], acc[j][i]);
  }

  int col = lane & 15;
  int rowbase = (lane >> 4) << 3;  // C-layout: lanes>=16 hold rows 8..15
#pragma unroll
  for (int j = 0; j < 4; ++j) {
    int cg = nt * 64 + j * 16 + col;
    float bvv = bias[cg];
#pragma unroll
    for (int i = 0; i < 4; ++i) {
#pragma unroll
      for (int r = 0; r < 8; ++r) {
        int R = mb * 64 + i * 16 + rowbase + r;
        float v = acc[j][i][r] + bvv;
        if (MODE == 2) v = fmaxf(v, 0.0f);
        size_t addr;
        if (MODE == 0) {
          int bb = R >> 11, l = R & (L_ - 1), h2 = cg >> 6, d = cg & (DH_ - 1);
          addr = (((size_t)bb * H_ + h2) * L_ + l) * DH_ + d;
        } else if (MODE == 1) {
          int bb = R >> 11, l = R & (L_ - 1), h2 = cg >> 6, d = cg & (DH_ - 1);
          addr = (((size_t)bb * H_ + h2) * DH_ + d) * L_ + l;
        } else {
          addr = (size_t)z * strideO + (size_t)R * N + cg;
        }
        outp[addr] = f2bf(v);
      }
    }
  }
}

// ---------------- flash attention over W2 logits ----------------
// per wave: one (b,h), one 16-row tile of L; online softmax over 64 s-blocks of 32.
// hbuf: [B*H, L, HID] bf16 ; W2t: [L, HID] bf16 ; Vt: [B*H, DH, L] bf16.
__global__ void __launch_bounds__(32)
attn_kernel(const u16* __restrict__ hbuf, const u16* __restrict__ W2t,
            const u16* __restrict__ Vt, const float* __restrict__ mask,
            float* __restrict__ out) {
  __shared__ __attribute__((aligned(16))) u16 Pt[16 * 32];
#if HAVE_ASYNC_LDS
  __shared__ __attribute__((aligned(16))) u16 Vst[2][64 * 32];
#endif
  int lane = threadIdx.x;
  int lt = blockIdx.x;   // L/16
  int bh = blockIdx.y;   // B*H
  int b = bh >> 4, hh = bh & 15;

  const u16* Ah = hbuf + ((size_t)bh * L_ + (size_t)lt * 16) * HID_;
  const u16* Vb = Vt + (size_t)bh * DH_ * L_;

  // Q-side A fragments (16 rows x K=128), loaded once, reused for all s-blocks.
  v16bf afrag[4];
#pragma unroll
  for (int kk = 0; kk < 4; ++kk) afrag[kk] = load_a_frag(Ah + kk * 32, HID_, lane);

  float mrow[8], srow[8];
  v8f o0 = {}, o1 = {}, o2 = {}, o3 = {};
#pragma unroll
  for (int r = 0; r < 8; ++r) { mrow[r] = -3.0e38f; srow[r] = 0.0f; }
  int col = lane & 15;
  int rowbase = (lane >> 4) << 3;

#if HAVE_ASYNC_LDS
  stage_v(Vb + 0, &Vst[0][0], lane);   // prologue: stage V-block 0
#endif

  for (int s0 = 0; s0 < L_; s0 += 32) {
#if HAVE_ASYNC_LDS
    int buf = (s0 >> 5) & 1;
    wait_async0();                      // current V block resident in LDS
    if (s0 + 32 < L_) stage_v(Vb + s0 + 32, &Vst[buf ^ 1][0], lane);  // overlap next copy
#endif
    // logits for 32 columns: two 16x16 C tiles, K=128 in 4 WMMA steps each
    v8f c0 = {}, c1 = {};
#pragma unroll
    for (int kk = 0; kk < 4; ++kk) {
      v16bf bq0 = load_b_frag(W2t + (size_t)s0 * HID_ + kk * 32, HID_, lane);
      c0 = WMMA_BF16(afrag[kk], bq0, c0);
      v16bf bq1 = load_b_frag(W2t + (size_t)(s0 + 16) * HID_ + kk * 32, HID_, lane);
      c1 = WMMA_BF16(afrag[kk], bq1, c1);
    }
    float pen0 = -1.0e6f * (1.0f - mask[b * L_ + s0 + col]);
    float pen1 = -1.0e6f * (1.0f - mask[b * L_ + s0 + 16 + col]);

    // online softmax: halves are row-disjoint, reduce within 16-lane groups
#pragma unroll
    for (int r = 0; r < 8; ++r) {
      c0[r] += pen0; c1[r] += pen1;
      float t = fmaxf(c0[r], c1[r]);
      t = fmaxf(t, __shfl_xor(t, 1, 32));
      t = fmaxf(t, __shfl_xor(t, 2, 32));
      t = fmaxf(t, __shfl_xor(t, 4, 32));
      t = fmaxf(t, __shfl_xor(t, 8, 32));
      float nm = fmaxf(mrow[r], t);
      float al = __expf(mrow[r] - nm);
      mrow[r] = nm;
      float p0 = __expf(c0[r] - nm);
      float p1 = __expf(c1[r] - nm);
      c0[r] = p0; c1[r] = p1;
      float ts = p0 + p1;
      ts += __shfl_xor(ts, 1, 32);
      ts += __shfl_xor(ts, 2, 32);
      ts += __shfl_xor(ts, 4, 32);
      ts += __shfl_xor(ts, 8, 32);
      srow[r] = srow[r] * al + ts;
      o0[r] *= al; o1[r] *= al; o2[r] *= al; o3[r] *= al;
    }

    // re-swizzle P: C layout -> A layout via LDS (16x32 bf16)
#pragma unroll
    for (int r = 0; r < 8; ++r) {
      Pt[(rowbase + r) * 32 + col]      = f2bf(c0[r]);
      Pt[(rowbase + r) * 32 + 16 + col] = f2bf(c1[r]);
    }
    __syncthreads();
    v16bf pa = load_a_frag(Pt, 32, lane);
    __syncthreads();

    // O += P @ V  (V^T rows = d, contiguous in s => clean B fragments)
#if HAVE_ASYNC_LDS
    const u16* Vs = &Vst[buf][0];
    v16bf bv0 = load_b_frag(Vs + (0 * 16) * 32, 32, lane);
    o0 = WMMA_BF16(pa, bv0, o0);
    v16bf bv1 = load_b_frag(Vs + (1 * 16) * 32, 32, lane);
    o1 = WMMA_BF16(pa, bv1, o1);
    v16bf bv2 = load_b_frag(Vs + (2 * 16) * 32, 32, lane);
    o2 = WMMA_BF16(pa, bv2, o2);
    v16bf bv3 = load_b_frag(Vs + (3 * 16) * 32, 32, lane);
    o3 = WMMA_BF16(pa, bv3, o3);
#else
    v16bf bv0 = load_b_frag(Vb + (size_t)(0 * 16) * L_ + s0, L_, lane);
    o0 = WMMA_BF16(pa, bv0, o0);
    v16bf bv1 = load_b_frag(Vb + (size_t)(1 * 16) * L_ + s0, L_, lane);
    o1 = WMMA_BF16(pa, bv1, o1);
    v16bf bv2 = load_b_frag(Vb + (size_t)(2 * 16) * L_ + s0, L_, lane);
    o2 = WMMA_BF16(pa, bv2, o2);
    v16bf bv3 = load_b_frag(Vb + (size_t)(3 * 16) * L_ + s0, L_, lane);
    o3 = WMMA_BF16(pa, bv3, o3);
#endif
  }

  // epilogue: normalize and scatter to [B, L, H*DH] f32
#pragma unroll
  for (int r = 0; r < 8; ++r) {
    float inv = 1.0f / srow[r];
    int l = lt * 16 + rowbase + r;
    size_t base = ((size_t)b * L_ + l) * (H_ * DH_) + (size_t)hh * DH_;
    out[base + 0  + col] = o0[r] * inv;
    out[base + 16 + col] = o1[r] * inv;
    out[base + 32 + col] = o2[r] * inv;
    out[base + 48 + col] = o3[r] * inv;
  }
}

// ---------------- host launcher ----------------

extern "C" void kernel_launch(void* const* d_in, const int* in_sizes, int n_in,
                              void* d_out, int out_size, void* d_ws, size_t ws_size,
                              hipStream_t stream) {
  (void)in_sizes; (void)n_in; (void)out_size; (void)ws_size;
  const float* X    = (const float*)d_in[0];
  const float* mask = (const float*)d_in[1];
  const float* Wx   = (const float*)d_in[2];
  const float* bx   = (const float*)d_in[3];
  const float* Wv   = (const float*)d_in[4];
  const float* bv   = (const float*)d_in[5];
  const float* W1   = (const float*)d_in[6];
  const float* b1   = (const float*)d_in[7];
  const float* W2   = (const float*)d_in[8];
  float* out = (float*)d_out;

  char* ws = (char*)d_ws;
  size_t off = 0;
  auto carve = [&](size_t bytes) -> char* {
    off = (off + 255) & ~(size_t)255;
    char* p = ws + off;
    off += bytes;
    return p;
  };
  u16* Xbf = (u16*)carve((size_t)B_ * L_ * DIM_ * 2);        // [B*L, DIM]
  u16* Wxt = (u16*)carve((size_t)DIM_ * H_ * DH_ * 2);       // [1024, 1024]^T
  u16* Wvt = (u16*)carve((size_t)DIM_ * H_ * DH_ * 2);
  u16* W1t = (u16*)carve((size_t)HID_ * DH_ * 2);            // [128, 64]
  u16* W2t = (u16*)carve((size_t)L_ * HID_ * 2);             // [2048, 128]
  u16* Xh  = (u16*)carve((size_t)B_ * H_ * L_ * DH_ * 2);    // [B,H,L,DH]
  u16* Vtb = (u16*)carve((size_t)B_ * H_ * DH_ * L_ * 2);    // [B,H,DH,L]
  u16* hb  = (u16*)carve((size_t)B_ * H_ * L_ * HID_ * 2);   // [B*H, L, HID]

  // 1) conversions
  {
    int n = B_ * L_ * DIM_;
    k_f32_to_bf16<<<(n + 255) / 256, 256, 0, stream>>>(X, Xbf, n);
  }
  k_transpose_bf16<<<(DIM_ * 1024 + 255) / 256, 256, 0, stream>>>(Wx, Wxt, DIM_, 1024);
  k_transpose_bf16<<<(DIM_ * 1024 + 255) / 256, 256, 0, stream>>>(Wv, Wvt, DIM_, 1024);
  k_transpose_bf16<<<(DH_ * HID_ + 255) / 256, 256, 0, stream>>>(W1, W1t, DH_, HID_);
  k_transpose_bf16<<<(HID_ * L_ + 255) / 256, 256, 0, stream>>>(W2, W2t, HID_, L_);

  // 2) Xh = X@Wx + bx   [4096 x 1024] (head-split store), 64x64 tile per wave
  gemm_wmma_k<0><<<dim3(16, 64, 1), 32, 0, stream>>>(Xbf, Wxt, bx, Xh,
                                                     B_ * L_, 1024, DIM_, 0, 0);
  // 3) V^T = (X@Wv + bv)^T per head
  gemm_wmma_k<1><<<dim3(16, 64, 1), 32, 0, stream>>>(Xbf, Wvt, bv, Vtb,
                                                     B_ * L_, 1024, DIM_, 0, 0);
  // 4) h = relu(Xh@W1 + b1), batched over B*H
  gemm_wmma_k<2><<<dim3(HID_ / 64, L_ / 64, B_ * H_), 32, 0, stream>>>(
      Xh, W1t, b1, hb, L_, HID_, DH_, (size_t)L_ * DH_, (size_t)L_ * HID_);

  // 5) flash attention
  attn_kernel<<<dim3(L_ / 16, B_ * H_), 32, 0, stream>>>(hb, W2t, Vtb, mask, out);
}
